// Attention_88510686036059
// MI455X (gfx1250) — matrix operand order repaired and checked by reading, not compiled, not run on hardware
//
#include <hip/hip_runtime.h>
#include <hip/hip_bf16.h>

typedef _Float16 v16h __attribute__((ext_vector_type(16)));
typedef float    v8f  __attribute__((ext_vector_type(8)));

#define AN 128   // agents
#define LN 32    // time steps
#define DN 256   // lstm dim
#define HN 8     // heads
#define HD 32    // head dim
#define BHN 64   // B*H
#define ATTN_ELEMS (64 * 128 * 128 * 32)  // B*H*A*A*L

static __device__ __forceinline__ v8f wmma_f16(v16h a, v16h b, v8f c) {
    return __builtin_amdgcn_wmma_f32_16x16x32_f16(false, a, false, b, (short)0, c,
                                                  false, false);
}

// ---------------------------------------------------------------------------
// Kernel 1: Q/K/V projections. X(32768x256) @ W(256x256), fp32->f16 convert.
// grid (256 row-blocks, 3 matrices, 4 col-groups of 64), block 256 (8 waves).
// q16/k16 layout: [b][h][l][a][hd]; vT16 layout: [b][h][a][hd][l].
// ---------------------------------------------------------------------------
__global__ __launch_bounds__(256) void proj_kernel(
    const float* __restrict__ X, const float* __restrict__ Wq,
    const float* __restrict__ Wk, const float* __restrict__ Wv,
    _Float16* __restrict__ q16, _Float16* __restrict__ k16,
    _Float16* __restrict__ vT16)
{
    __shared__ _Float16 Wt[64 * 256];   // W^T slice: [col_local][k], 32 KB

    const int which = blockIdx.y;       // 0=q, 1=k, 2=v
    const float* W = (which == 0) ? Wq : (which == 1) ? Wk : Wv;
    const int cg = blockIdx.z;          // column group (64 cols each)
    const int tid = threadIdx.x;

    for (int idx = tid; idx < 64 * 256; idx += 256) {
        int c = idx & 63, k = idx >> 6;
        Wt[c * 256 + k] = (_Float16)W[k * 256 + cg * 64 + c];
    }
    __syncthreads();

    const int wave = tid >> 5, lane = tid & 31;
    const int hl = lane >> 4, ln = lane & 15;
    const int baseRow = (blockIdx.x * 8 + wave) * 16;
    const int rowA = baseRow + ln;

    // A fragments: 16 rows x 256 K, split into 8 K-chunks of 32.
    // A layout: lane holds K contiguous [16*hl, 16*hl+16).
    v16h afrag[8];
    #pragma unroll
    for (int kc = 0; kc < 8; ++kc) {
        const float* p0 = X + rowA * 256 + kc * 32 + hl * 16;
        v16h a;
        #pragma unroll
        for (int e = 0; e < 8; ++e) {
            a[e]     = (_Float16)p0[e];
            a[8 + e] = (_Float16)p0[8 + e];
        }
        afrag[kc] = a;
    }

    #pragma unroll
    for (int nt = 0; nt < 4; ++nt) {
        v8f acc = {};
        #pragma unroll
        for (int kc = 0; kc < 8; ++kc) {
            // B layout: lane = column n, K contiguous [16*hl, 16*hl+16)
            v16h b = *(const v16h*)&Wt[(nt * 16 + ln) * 256 + kc * 32 + 16 * hl];
            acc = wmma_f16(afrag[kc], b, acc);
        }
        const int col = cg * 64 + nt * 16 + ln;
        const int h = col >> 5, d = col & 31;
        #pragma unroll
        for (int r = 0; r < 8; ++r) {
            const int row = baseRow + r + 8 * hl;     // C/D: M = r + 8*half
            const int l = row >> 10, ba = row & 1023;
            const int b_ = ba >> 7, a_ = ba & 127;
            const int bh = b_ * HN + h;
            const _Float16 val = (_Float16)acc[r];
            if (which == 2) {
                vT16[((bh * AN + a_) * HD + d) * LN + l] = val;
            } else {
                const int idx = ((bh * LN + l) * AN + a_) * HD + d;
                if (which == 0) q16[idx] = val; else k16[idx] = val;
            }
        }
    }
}

// ---------------------------------------------------------------------------
// Kernel 2: scores + softmax over L. One wave per (b,h, i-tile, j-tile).
// scores[i,j] = k_agents(16x32) @ q_agents^T(32x16) per l; softmax over l.
// Writes fp32 attn directly to d_out:  [(bh*A + i)*A + j]*L + l.
// ---------------------------------------------------------------------------
__global__ __launch_bounds__(32) void attn_kernel(
    const _Float16* __restrict__ q16, const _Float16* __restrict__ k16,
    float* __restrict__ attn)
{
    __shared__ float sc[LN * 256];   // [l][m*16+n], 32 KB
    const int bh = blockIdx.x, it = blockIdx.y, jt = blockIdx.z;
    const int lane = threadIdx.x, hl = lane >> 4, ln = lane & 15;

    for (int l = 0; l < LN; ++l) {
        const _Float16* kp = k16 + ((bh * LN + l) * AN + it * 16 + ln) * HD;
        v16h a;
        #pragma unroll
        for (int e = 0; e < 8; ++e) {
            a[e]     = kp[hl * 16 + e];
            a[8 + e] = kp[hl * 16 + 8 + e];
        }
        v16h b = *(const v16h*)(q16 + ((bh * LN + l) * AN + jt * 16 + ln) * HD
                                + 16 * hl);
        v8f acc = {};
        acc = wmma_f16(a, b, acc);
        #pragma unroll
        for (int r = 0; r < 8; ++r)
            sc[l * 256 + (r + 8 * hl) * 16 + ln] = acc[r] * 0.0625f; // 1/sqrt(D)
    }
    __syncthreads();

    #pragma unroll
    for (int t = 0; t < 8; ++t) {
        const int p = t * 32 + lane;          // (m,n) pair, 0..255
        const int m = p >> 4, n = p & 15;
        float va[LN];
        #pragma unroll
        for (int l = 0; l < LN; ++l) va[l] = sc[l * 256 + p];
        float mx = va[0];
        #pragma unroll
        for (int l = 1; l < LN; ++l) mx = fmaxf(mx, va[l]);
        float s = 0.0f;
        #pragma unroll
        for (int l = 0; l < LN; ++l) { va[l] = __expf(va[l] - mx); s += va[l]; }
        const float inv = 1.0f / s;
        const int i = it * 16 + m, j = jt * 16 + n;
        float* op = attn + ((bh * AN + i) * AN + j) * LN;
        #pragma unroll
        for (int l4 = 0; l4 < 8; ++l4) {
            float4 o = make_float4(va[l4 * 4 + 0] * inv, va[l4 * 4 + 1] * inv,
                                   va[l4 * 4 + 2] * inv, va[l4 * 4 + 3] * inv);
            *reinterpret_cast<float4*>(op + l4 * 4) = o;
        }
    }
}

// ---------------------------------------------------------------------------
// Kernel 3: out = attn(128x32) @ v(32x32) per (b,h,i), fused with the GAT
// dot-products against Watt. Deterministic (no atomics): per-head partials
// P1[b,h,i,j] (Watt[0:256] term) and P2[b,h,j,i]->stored so K4 reads [b,h,i,j].
// ---------------------------------------------------------------------------
__global__ __launch_bounds__(32) void out_kernel(
    const float* __restrict__ attn, const _Float16* __restrict__ vT16,
    const float* __restrict__ Watt,
    float* __restrict__ P1, float* __restrict__ P2)
{
    __shared__ float T[AN * HD];   // out tile [j][d], 16 KB
    const int bh = blockIdx.x, i = blockIdx.y;
    const int lane = threadIdx.x, hl = lane >> 4, ln = lane & 15;

    v16h vfrag[2];
    #pragma unroll
    for (int dt = 0; dt < 2; ++dt)
        vfrag[dt] = *(const v16h*)(vT16 + ((bh * AN + i) * HD + dt * 16 + ln) * LN
                                   + 16 * hl);

    #pragma unroll
    for (int jtile = 0; jtile < 8; ++jtile) {
        const float* ap = attn + ((bh * AN + i) * AN + jtile * 16 + ln) * LN;
        v16h a;
        #pragma unroll
        for (int e = 0; e < 8; ++e) {
            a[e]     = (_Float16)ap[hl * 16 + e];
            a[8 + e] = (_Float16)ap[hl * 16 + 8 + e];
        }
        #pragma unroll
        for (int dt = 0; dt < 2; ++dt) {
            v8f acc = {};
            acc = wmma_f16(a, vfrag[dt], acc);
            #pragma unroll
            for (int r = 0; r < 8; ++r)
                T[(jtile * 16 + r + 8 * hl) * HD + dt * 16 + ln] = acc[r];
        }
    }
    __syncthreads();

    const int h = bh & 7;
    float w1[HD], w2[HD];
    #pragma unroll
    for (int d = 0; d < HD; ++d) {
        w1[d] = Watt[h * HD + d];
        w2[d] = Watt[DN + h * HD + d];
    }
    #pragma unroll
    for (int t = 0; t < 4; ++t) {
        const int J = t * 32 + lane;
        float s1 = 0.0f, s2 = 0.0f;
        #pragma unroll
        for (int d = 0; d < HD; ++d) {
            const float v = T[J * HD + d];
            s1 += v * w1[d];
            s2 += v * w2[d];
        }
        P1[(bh * AN + i) * AN + J] = s1;    // contributes to logits[b,i,J]
        P2[(bh * AN + J) * AN + i] = s2;    // contributes to logits[b,J,i]
    }
}

// ---------------------------------------------------------------------------
// Kernel 4: logits = sum_h (P1 + P2) + b_att; weight = tanh(logits+0.5)*mask.
// ---------------------------------------------------------------------------
__global__ __launch_bounds__(256) void gat_kernel(
    const float* __restrict__ P1, const float* __restrict__ P2,
    const float* __restrict__ b_att, float* __restrict__ wout)
{
    const int idx = blockIdx.x * 256 + threadIdx.x;   // (b,i,j)
    const int j = idx & 127, i = (idx >> 7) & 127, b = idx >> 14;
    float acc = b_att[0] + 0.5f;
    #pragma unroll
    for (int h = 0; h < HN; ++h) {
        const int o = (((b * HN + h) * AN + i) * AN + j);
        acc += P1[o] + P2[o];
    }
    const float w = tanhf(acc);
    wout[idx] = (i == j) ? 0.0f : w;
}

extern "C" void kernel_launch(void* const* d_in, const int* in_sizes, int n_in,
                              void* d_out, int out_size, void* d_ws, size_t ws_size,
                              hipStream_t stream) {
    const float* X     = (const float*)d_in[0];   // output (L, B*A, D)
    const float* Wk    = (const float*)d_in[1];
    const float* Wq    = (const float*)d_in[2];
    const float* Wv    = (const float*)d_in[3];
    const float* Watt  = (const float*)d_in[4];
    const float* b_att = (const float*)d_in[5];
    // d_in[6] = agent_num (compile-time constant AN here)

    float* attn   = (float*)d_out;                 // (B,H,A,A,L) fp32
    float* weight = attn + (size_t)ATTN_ELEMS;     // (B,A,A) fp32

    char* ws = (char*)d_ws;
    _Float16* q16  = (_Float16*)(ws);                           // 16 MB
    _Float16* k16  = (_Float16*)(ws + ((size_t)16 << 20));      // 16 MB
    _Float16* vT16 = (_Float16*)(ws + ((size_t)32 << 20));      // 16 MB
    float*    P1   = (float*)   (ws + ((size_t)48 << 20));      //  4 MB
    float*    P2   = (float*)   (ws + ((size_t)52 << 20));      //  4 MB

    proj_kernel<<<dim3(256, 3, 4), 256, 0, stream>>>(X, Wq, Wk, Wv, q16, k16, vT16);
    attn_kernel<<<dim3(BHN, 8, 8), 32, 0, stream>>>(q16, k16, attn);
    out_kernel <<<dim3(BHN, AN),   32, 0, stream>>>(attn, vT16, Watt, P1, P2);
    gat_kernel <<<dim3(512),      256, 0, stream>>>(P1, P2, b_att, weight);
}
